// GraphAttentionModule_37203006718541
// MI455X (gfx1250) — matrix operand order repaired and checked by reading, not compiled, not run on hardware
//
#include <hip/hip_runtime.h>
#include <hip/hip_bf16.h>

// GAT module, fused per-timestep kernel for gfx1250 (MI455X).
// Precision: bf16 WMMA (16x16x32, f32 accumulate) for both GEMMs; f32 softmax.
// Roofline: 16.8 GFLOP vs 82 MB (AI ~205 FLOP/B) -> matrix cores required; bf16 path.

#define NPAD   112   // 100 nodes / per-head OUT padded to 7*16
#define KPAD   128   // K (=100) padded to 4*32
#define STRIDE 136   // LDS row stride in bf16 elems: 272B, 16B-aligned, bank-skewed

typedef __attribute__((ext_vector_type(16))) __bf16 v16bf;
typedef __attribute__((ext_vector_type(8)))  float  v8f;

union Frag { uint4 u4[2]; v16bf v; };

__device__ __forceinline__ float lrelu(float x) { return x > 0.0f ? x : 0.2f * x; }

__device__ __forceinline__ v8f wmma_bf16(Frag a, Frag b, v8f c) {
  // (neg_a, A, neg_b, B, c_mod, C, reuse_a, reuse_b)
  return __builtin_amdgcn_wmma_f32_16x16x32_bf16(false, a.v, false, b.v,
                                                 (short)0, c, false, false);
}

// ---- prep: W [400,100] f32 -> Wb [448,128] bf16 (head stride 112, zero padded)
__global__ void prep_w_kernel(const float* __restrict__ W, __bf16* __restrict__ Wb) {
  int idx = blockIdx.x * 256 + threadIdx.x;
  if (idx >= 448 * KPAD) return;
  int r = idx >> 7;          // padded row  (h*112 + o)
  int f = idx & (KPAD - 1);  // padded K
  int h = r / NPAD;
  int o = r - h * NPAD;
  float v = 0.0f;
  if (o < 100 && f < 100) v = W[(h * 100 + o) * 100 + f];
  Wb[idx] = (__bf16)v;
}

// LDS layout
static constexpr int HT_BYTES  = 448 * STRIDE * 2;          // H^T  [448][STRIDE] bf16
static constexpr int AB_BYTES  = NPAD * STRIDE * 2;         // X / alpha [112][STRIDE] bf16
static constexpr int F32_BYTES = 4 * (4 * NPAD) * 4 + 16;   // attS, attD, aS, aD, smax
static constexpr int SMEM_BYTES = HT_BYTES + AB_BYTES + F32_BYTES; // 159,504 (16B mult)

__global__ __launch_bounds__(256)
void gat_fused(const float* __restrict__ A, const __bf16* __restrict__ Wb,
               const float* __restrict__ attSrcG, const float* __restrict__ attDstG,
               const float* __restrict__ biasG, float* __restrict__ out) {
  __shared__ __align__(16) unsigned char smem[SMEM_BYTES];
  __bf16* Ht   = (__bf16*)smem;                         // [448][STRIDE]
  __bf16* Ab   = (__bf16*)(smem + HT_BYTES);            // X (phase 1), alpha (phase 3+)
  float*  attS = (float*)(smem + HT_BYTES + AB_BYTES);  // [4][112]
  float*  attD = attS + 4 * NPAD;
  float*  aS   = attD + 4 * NPAD;
  float*  aD   = aS   + 4 * NPAD;
  float*  smax = aD   + 4 * NPAD;                       // [4]

  const int tid  = threadIdx.x;
  const int lane = tid & 31;
  const int wave = tid >> 5;
  const int t    = blockIdx.x;
  const float* At = A + (size_t)t * 10000;

  // ---- zero-init all LDS (padding must be true zeros: WMMA propagates NaN even *0)
  for (int i = tid; i < SMEM_BYTES / 16; i += 256)
    ((int4*)smem)[i] = make_int4(0, 0, 0, 0);
  __syncthreads();

  // ---- load X = A_t^T as bf16 into Ab (row n holds K-contiguous data = B-matrix layout)
  for (int idx = tid; idx < 10000; idx += 256) {
    int f = idx / 100, n = idx - f * 100;       // A_t[f][n], coalesced over n
    Ab[n * STRIDE + f] = (__bf16)At[idx];
  }
  // attention vectors to LDS (f32, padded)
  for (int idx = tid; idx < 4 * NPAD; idx += 256) {
    int h = idx / NPAD, o = idx - h * NPAD;
    float vs = 0.0f, vd = 0.0f;
    if (o < 100) { vs = attSrcG[h * 100 + o]; vd = attDstG[h * 100 + o]; }
    attS[idx] = vs; attD[idx] = vd;
  }
  __syncthreads();

  // ---- GEMM1: H^T[r][n] = sum_f Wb[r][f] * X[n][f]   (M=448, N=112, K=128)
  const int kbA = (lane >> 4) * 8;    // A-frag K sub-offset per lane half
  const int kbB = (lane >> 4) * 8;    // B-frag K sub-offset per lane half
  #pragma unroll
  for (int im = 0; im < 4; ++im) {
    int mt = wave + im * 8;           // wave-uniform -> EXEC stays all-ones
    if (mt < 28) {
      Frag afr[4];
      const __bf16* wrow = Wb + (mt * 16 + (lane & 15)) * KPAD + kbA;
      #pragma unroll
      for (int ks = 0; ks < 4; ++ks) {
        afr[ks].u4[0] = *(const uint4*)(wrow + ks * 32);
        afr[ks].u4[1] = *(const uint4*)(wrow + ks * 32 + 16);
      }
      for (int nt = 0; nt < 7; ++nt) {
        // batch all B-fragment loads, then issue the 4 WMMAs back-to-back
        const __bf16* xrow = Ab + (nt * 16 + (lane & 15)) * STRIDE + kbB;
        Frag bfr[4];
        #pragma unroll
        for (int ks = 0; ks < 4; ++ks) {
          bfr[ks].u4[0] = *(const uint4*)(xrow + ks * 32);
          bfr[ks].u4[1] = *(const uint4*)(xrow + ks * 32 + 16);
        }
        v8f acc = {0.f, 0.f, 0.f, 0.f, 0.f, 0.f, 0.f, 0.f};
        #pragma unroll
        for (int ks = 0; ks < 4; ++ks) acc = wmma_bf16(afr[ks], bfr[ks], acc);
        int r0 = mt * 16 + ((lane >> 4) * 8);
        int c  = nt * 16 + (lane & 15);
        #pragma unroll
        for (int v = 0; v < 8; ++v)
          Ht[(r0 + v) * STRIDE + c] = (__bf16)acc[v];
      }
    }
  }
  __syncthreads();

  // ---- attention logits: a_s[h][n] = sum_o H^T[h*112+o][n]*att_src[h][o]  (f32)
  for (int idx = tid; idx < 4 * 100; idx += 256) {
    int h = idx / 100, n = idx - h * 100;
    const __bf16* col = Ht + (h * NPAD) * STRIDE + n;
    const float* sv = attS + h * NPAD;
    const float* dv = attD + h * NPAD;
    float ss = 0.0f, sd = 0.0f;
    for (int o = 0; o < 100; ++o) {
      float hv = (float)col[o * STRIDE];
      ss = fmaf(hv, sv[o], ss);
      sd = fmaf(hv, dv[o], sd);
    }
    aS[h * NPAD + n] = ss;
    aD[h * NPAD + n] = sd;
  }
  __syncthreads();
  // lrelu is monotone increasing => max_j lrelu(a_d+a_s[j]) = lrelu(a_d + max_j a_s[j])
  if (tid < 4) {
    float m = -1e30f;
    for (int j = 0; j < 100; ++j) m = fmaxf(m, aS[tid * NPAD + j]);
    smax[tid] = m;
  }

  // ---- per-head: softmax rows -> alpha bf16 (reusing Ab), GEMM2 accumulate in regs
  v8f acc2[7];
  const v8f vzero = {0.f, 0.f, 0.f, 0.f, 0.f, 0.f, 0.f, 0.f};
  #pragma unroll
  for (int i = 0; i < 7; ++i) acc2[i] = vzero;

  for (int h = 0; h < 4; ++h) {
    __syncthreads();  // prior GEMM2 reads of Ab done (h=0: smax visible)
    for (int i = wave; i < NPAD; i += 8) {   // wave-per-row softmax
      int j0 = lane, j1 = lane + 32, j2 = lane + 64, j3 = lane + 96;
      if (i < 100) {
        float adi = aD[h * NPAD + i];
        float mi  = lrelu(adi + smax[h]);
        float e0 = 0.f, e1 = 0.f, e2 = 0.f, e3 = 0.f;
        if (j0 < 100) e0 = __expf(lrelu(adi + aS[h * NPAD + j0]) - mi);
        if (j1 < 100) e1 = __expf(lrelu(adi + aS[h * NPAD + j1]) - mi);
        if (j2 < 100) e2 = __expf(lrelu(adi + aS[h * NPAD + j2]) - mi);
        if (j3 < 100) e3 = __expf(lrelu(adi + aS[h * NPAD + j3]) - mi);
        float s = e0 + e1 + e2 + e3;
        #pragma unroll
        for (int off = 16; off > 0; off >>= 1) s += __shfl_xor(s, off, 32);
        float inv = 1.0f / s;
        Ab[i * STRIDE + j0] = (__bf16)(e0 * inv);
        Ab[i * STRIDE + j1] = (__bf16)(e1 * inv);
        Ab[i * STRIDE + j2] = (__bf16)(e2 * inv);
        Ab[i * STRIDE + j3] = (__bf16)(e3 * inv);
      } else {
        Ab[i * STRIDE + j0] = (__bf16)0.0f;
        Ab[i * STRIDE + j1] = (__bf16)0.0f;
        Ab[i * STRIDE + j2] = (__bf16)0.0f;
        Ab[i * STRIDE + j3] = (__bf16)0.0f;
      }
    }
    __syncthreads();
    // GEMM2: out^T[o][i] += sum_j H^T_h[o][j] * alpha[i][j]   (A=Ht rows, B=Ab rows)
    #pragma unroll
    for (int it = 0; it < 7; ++it) {
      int tt = wave + it * 8;          // wave-uniform tile id over 7x7 grid
      if (tt < 49) {
        int mt = tt / 7, nt = tt - mt * 7;
        const __bf16* arow = Ht + (h * NPAD + mt * 16 + (lane & 15)) * STRIDE + kbA;
        const __bf16* brow = Ab + (nt * 16 + (lane & 15)) * STRIDE + kbB;
        // batch all 8 fragment loads (one clause, single wait), then 4 WMMAs
        Frag a[4], b[4];
        #pragma unroll
        for (int ks = 0; ks < 4; ++ks) {
          a[ks].u4[0] = *(const uint4*)(arow + ks * 32);
          a[ks].u4[1] = *(const uint4*)(arow + ks * 32 + 16);
        }
        #pragma unroll
        for (int ks = 0; ks < 4; ++ks) {
          b[ks].u4[0] = *(const uint4*)(brow + ks * 32);
          b[ks].u4[1] = *(const uint4*)(brow + ks * 32 + 16);
        }
        #pragma unroll
        for (int ks = 0; ks < 4; ++ks)
          acc2[it] = wmma_bf16(a[ks], b[ks], acc2[it]);
      }
    }
  }

  // ---- epilogue: mean over heads, +bias, lrelu, +I; output is already [t][o][i]
  #pragma unroll
  for (int it = 0; it < 7; ++it) {
    int tt = wave + it * 8;
    if (tt < 49) {
      int mt = tt / 7, nt = tt - mt * 7;
      int i = nt * 16 + (lane & 15);
      if (i < 100) {
        #pragma unroll
        for (int v = 0; v < 8; ++v) {
          int o = mt * 16 + ((lane >> 4) * 8) + v;
          if (o < 100) {
            float val = 0.25f * acc2[it][v] + biasG[o];
            val = lrelu(val);
            if (o == i) val += 1.0f;
            out[(size_t)t * 10000 + o * 100 + i] = val;
          }
        }
      }
    }
  }
}

extern "C" void kernel_launch(void* const* d_in, const int* in_sizes, int n_in,
                              void* d_out, int out_size, void* d_ws, size_t ws_size,
                              hipStream_t stream) {
  const float* A       = (const float*)d_in[0];
  const float* W       = (const float*)d_in[1];
  const float* att_src = (const float*)d_in[2];
  const float* att_dst = (const float*)d_in[3];
  const float* bias    = (const float*)d_in[4];
  float* out = (float*)d_out;
  __bf16* Wb = (__bf16*)d_ws;   // 448*128*2 = 114,688 B of scratch

  int T = in_sizes[0] / 10000;  // 1024

  prep_w_kernel<<<(448 * KPAD + 255) / 256, 256, 0, stream>>>(W, Wb);
  gat_fused<<<T, 256, 0, stream>>>(A, Wb, att_src, att_dst, bias, out);
}
